// FlashMoeAttention_11433202942603
// MI455X (gfx1250) — compile-verified, matrix operand order checked by reading, not run on hardware
//
#include <hip/hip_runtime.h>
#include <hip/hip_bf16.h>

// ---------------- constants ----------------
#define B_   4
#define S_   2048
#define D_   1024
#define E_   8
#define H_   16
#define HD_  64
#define TOK_ (B_ * S_)     // 8192
#define N3D_ (3 * D_)      // 3072

typedef __attribute__((ext_vector_type(16))) __bf16 v16bf;
typedef __attribute__((ext_vector_type(8)))  float  v8f;

#define WMMA_BF16(A, B, C) \
  __builtin_amdgcn_wmma_f32_16x16x32_bf16(false, (A), false, (B), (short)0, (C), false, false)

// CDNA5 async global->LDS path (ASYNCcnt), compile-safe guarded.
// Toolchain signature (from clang diagnostic): params are int4-vector pointers,
// global side in AS1 (printed as __device__), LDS side in AS3.
#if __has_builtin(__builtin_amdgcn_global_load_async_to_lds_b128)
#define HAVE_ASYNC 1
typedef int v4i_t __attribute__((vector_size(16)));
typedef __attribute__((address_space(1))) v4i_t as1_v4i;
typedef __attribute__((address_space(3))) v4i_t as3_v4i;
#else
#define HAVE_ASYNC 0
#endif

__device__ __forceinline__ void async_wait_all() {
#if HAVE_ASYNC
#if __has_builtin(__builtin_amdgcn_s_wait_asynccnt)
  __builtin_amdgcn_s_wait_asynccnt(0);
#else
  asm volatile("s_wait_asynccnt 0x0" ::: "memory");
#endif
#endif
}

// ---------------- helpers ----------------
__device__ __forceinline__ unsigned short f2bf(float x) {
  unsigned u = __float_as_uint(x);
  unsigned r = (u + 0x7FFFu + ((u >> 16) & 1u)) >> 16;
  return (unsigned short)r;
}

union BFfrag { v16bf v; unsigned int u[8]; };

// Build a 16x32 bf16 A/B fragment from an LDS tile stored row-major along K
// (A: [m][k], B: [n][k] i.e. B^T).  ISA 7.12.2 16-bit layout:
// lane m = lane&15; VGPR v holds K = 2v + (v>=4?8:0) + (lane>=16?8:0), pairs packed.
__device__ __forceinline__ v16bf load_frag_rowK(const unsigned short* p, int row0,
                                                int ld, int k0, int lane) {
  BFfrag f;
  int m  = lane & 15;
  int hi = lane >> 4;
  const unsigned short* rp = p + (row0 + m) * ld + k0 + hi * 8;
#pragma unroll
  for (int v = 0; v < 8; ++v) {
    int kk = 2 * v + ((v >= 4) ? 8 : 0);
    f.u[v] = *(const unsigned int*)(rp + kk);   // even offset -> 4B aligned
  }
  return f.v;
}

__device__ __forceinline__ v8f vzero8() {
  v8f z;
#pragma unroll
  for (int i = 0; i < 8; ++i) z[i] = 0.0f;
  return z;
}

// Stage a tile of contiguous rows (ROWS x UPR*8 ushorts) global -> LDS in 16B units.
// Uses CDNA5 async-to-LDS when available, else vector b128 copies.
template <int ROWS, int UPR, int LD, int NTHR>
__device__ __forceinline__ void stage_tile(unsigned short* dst,
                                           const unsigned short* src,
                                           size_t src_ld, int tid) {
#pragma unroll
  for (int u = tid; u < ROWS * UPR; u += NTHR) {
    int r = u / UPR, c = (u % UPR) * 8;
    const unsigned short* g = src + (size_t)r * src_ld + c;
    unsigned short* l = dst + r * LD + c;
#if HAVE_ASYNC
    __builtin_amdgcn_global_load_async_to_lds_b128(
        (as1_v4i*)(void*)g, (as3_v4i*)(void*)l, 0, 0);
#else
    *(uint4*)l = *(const uint4*)g;
#endif
  }
}

#define LDK 40   // k-tile LDS row (32 data + pad), 80B: 16B aligned rows
#define LDA 72   // attention tile LDS row (64 data + pad), 144B: 16B aligned rows

// ---------------- kernel 0a: fp32 -> bf16 (vectorized) ----------------
__global__ void cvt_bf16_kernel(const float* __restrict__ in,
                                unsigned short* __restrict__ out, int n4) {
  int i = blockIdx.x * blockDim.x + threadIdx.x;
  int stride = gridDim.x * blockDim.x;
  for (; i < n4; i += stride) {
    float4 f = ((const float4*)in)[i];
    ushort4 o;
    o.x = f2bf(f.x); o.y = f2bf(f.y); o.z = f2bf(f.z); o.w = f2bf(f.w);
    ((ushort4*)out)[i] = o;
  }
}

// ---------------- kernel 0b: fp32 (K x N) -> bf16 transposed (N x K) ----------------
// grid (K/32, N/32, batch), block 256
__global__ void cvt_bf16_transpose_kernel(const float* __restrict__ in,
                                          unsigned short* __restrict__ out,
                                          int K, int N) {
  __shared__ float tile[32][33];
  in  += (size_t)blockIdx.z * K * N;
  out += (size_t)blockIdx.z * K * N;
  const int k0 = blockIdx.x * 32;
  const int n0 = blockIdx.y * 32;
  const int tx = threadIdx.x & 31, ty = threadIdx.x >> 5;
#pragma unroll
  for (int i = ty; i < 32; i += 8)
    tile[i][tx] = in[(size_t)(k0 + i) * N + n0 + tx];   // coalesced read
  __syncthreads();
#pragma unroll
  for (int i = ty; i < 32; i += 8)
    out[(size_t)(n0 + i) * K + k0 + tx] = f2bf(tile[tx][i]);  // coalesced write
}

// ---------------- kernel 1: QKV GEMM ----------------
// grid (TOK/64, 3072/64), block 128 (4 waves); wave w -> rows [w*16, w*16+16)
// V is written TRANSPOSED: vtb[b][h][hd][s]
__global__ void qkv_gemm_kernel(const unsigned short* __restrict__ xb,
                                const unsigned short* __restrict__ wT,  // [N3D][D]
                                const float* __restrict__ bias,
                                unsigned short* __restrict__ qb,
                                unsigned short* __restrict__ kb,
                                unsigned short* __restrict__ vtb) {
  __shared__ __align__(16) unsigned short sx[64 * LDK];  // x tile  [m][k]
  __shared__ __align__(16) unsigned short sw[64 * LDK];  // W^T tile [n][k]
  const int row0 = blockIdx.x * 64;
  const int col0 = blockIdx.y * 64;
  const int tid  = threadIdx.x;
  const int lane = tid & 31;
  const int wv   = tid >> 5;
  const int hi8  = (lane >> 4) * 8;
  const int ln   = lane & 15;

  v8f acc[4];
#pragma unroll
  for (int t = 0; t < 4; ++t) acc[t] = vzero8();

  for (int k0 = 0; k0 < D_; k0 += 32) {
    __syncthreads();
    stage_tile<64, 4, LDK, 128>(sx, xb + (size_t)row0 * D_ + k0, D_, tid);
    stage_tile<64, 4, LDK, 128>(sw, wT + (size_t)col0 * D_ + k0, D_, tid);
    async_wait_all();
    __syncthreads();

    v16bf a = load_frag_rowK(sx, wv * 16, LDK, 0, lane);
#pragma unroll
    for (int t = 0; t < 4; ++t) {
      v16bf b = load_frag_rowK(sw, t * 16, LDK, 0, lane);
      acc[t] = WMMA_BF16(a, b, acc[t]);
    }
  }

  // epilogue: +bias, scatter to q/k (row-major heads) and v (transposed heads)
#pragma unroll
  for (int t = 0; t < 4; ++t) {
#pragma unroll
    for (int i = 0; i < 8; ++i) {
      int col = col0 + t * 16 + ln;
      int row = row0 + wv * 16 + i + hi8;
      float val = acc[t][i] + bias[col];
      int part = col >> 10, rem = col & 1023;
      int h = rem >> 6, dd = rem & 63;
      int b = row >> 11, s = row & 2047;
      unsigned short bv = f2bf(val);
      if (part == 2) {
        vtb[((size_t)((b * H_ + h) * HD_ + dd)) * S_ + s] = bv;   // [b][h][d][s]
      } else {
        size_t dst = ((size_t)((b * H_ + h) * S_ + s)) * HD_ + dd;
        if (part == 0) qb[dst] = bv; else kb[dst] = bv;
      }
    }
  }
}

// ---------------- kernel 2: gate softmax + top-2 ----------------
__global__ void gate_topk_kernel(const float* __restrict__ x,
                                 const float* __restrict__ Wg,
                                 const float* __restrict__ bg,
                                 float* __restrict__ comb) {
  const int tid  = threadIdx.x;
  const int lane = tid & 31;
  const int wv   = tid >> 5;
  const int token = blockIdx.x * 8 + wv;
  if (token >= TOK_) return;

  float acc[E_];
#pragma unroll
  for (int e = 0; e < E_; ++e) acc[e] = 0.0f;
  const float* xr = x + (size_t)token * D_;
  for (int d = lane; d < D_; d += 32) {
    float xv = xr[d];
#pragma unroll
    for (int e = 0; e < E_; ++e) acc[e] += xv * Wg[d * E_ + e];
  }
#pragma unroll
  for (int e = 0; e < E_; ++e) {
#pragma unroll
    for (int m = 16; m >= 1; m >>= 1) acc[e] += __shfl_xor(acc[e], m, 32);
  }
  if (lane == 0) {
    float l[E_], g[E_];
    float mx = -1e30f;
#pragma unroll
    for (int e = 0; e < E_; ++e) { l[e] = acc[e] + bg[e]; mx = fmaxf(mx, l[e]); }
    float den = 0.0f;
#pragma unroll
    for (int e = 0; e < E_; ++e) { g[e] = __expf(l[e] - mx); den += g[e]; }
    float inv = 1.0f / den;
    int i1 = 0; float b1 = -1e30f;
#pragma unroll
    for (int e = 0; e < E_; ++e) if (l[e] > b1) { b1 = l[e]; i1 = e; }
    int i2 = -1; float b2 = -1e30f;
#pragma unroll
    for (int e = 0; e < E_; ++e) if (e != i1 && l[e] > b2) { b2 = l[e]; i2 = e; }
#pragma unroll
    for (int e = 0; e < E_; ++e)
      comb[(size_t)token * E_ + e] = (e == i1 || e == i2) ? g[e] * inv : 0.0f;
  }
}

// ---------------- kernel 3: flash attention ----------------
// grid (S/64, B*H), block 128 (4 waves); wave owns 16 q-rows
__global__ void flash_attn_kernel(const unsigned short* __restrict__ qb,
                                  const unsigned short* __restrict__ kb,
                                  const unsigned short* __restrict__ vtb,
                                  float* __restrict__ out) {
  __shared__ __align__(16) unsigned short sq[64 * LDA];   // [m][d]
  __shared__ __align__(16) unsigned short sk[64 * LDA];   // [key][d]  (B^T for QK^T)
  __shared__ __align__(16) unsigned short svt[64 * LDA];  // [d][key]  (B^T for P@V)
  __shared__ __align__(16) unsigned short sp[4][16 * LDA]; // per-wave P strip

  const int bh = blockIdx.y;
  const int q0 = blockIdx.x * 64;
  const int tid  = threadIdx.x;
  const int lane = tid & 31;
  const int wv   = tid >> 5;
  const int hi8  = (lane >> 4) * 8;
  const int ln   = lane & 15;

  const unsigned short* qbase = qb  + (size_t)bh * S_ * HD_;
  const unsigned short* kbase = kb  + (size_t)bh * S_ * HD_;
  const unsigned short* vtbase = vtb + (size_t)bh * S_ * HD_;  // [d][s]

  stage_tile<64, 8, LDA, 128>(sq, qbase + (size_t)q0 * HD_, HD_, tid);
  async_wait_all();
  __syncthreads();
  v16bf aq0 = load_frag_rowK(sq, wv * 16, LDA, 0, lane);
  v16bf aq1 = load_frag_rowK(sq, wv * 16, LDA, 32, lane);

  float M[8], L[8];
  v8f o[4];
#pragma unroll
  for (int i = 0; i < 8; ++i) { M[i] = -1e30f; L[i] = 0.0f; }
#pragma unroll
  for (int t = 0; t < 4; ++t) o[t] = vzero8();

  const float scale = 0.125f;  // 1/sqrt(64)

  for (int kv0 = 0; kv0 < S_; kv0 += 64) {
    __syncthreads();
    stage_tile<64, 8, LDA, 128>(sk,  kbase  + (size_t)kv0 * HD_, HD_, tid);  // [key][d]
    stage_tile<64, 8, LDA, 128>(svt, vtbase + kv0,               S_,  tid);  // [d][key]
    async_wait_all();
    __syncthreads();

    // S = Q K^T
    v8f st[4];
#pragma unroll
    for (int t = 0; t < 4; ++t) {
      st[t] = vzero8();
      v16bf bk0 = load_frag_rowK(sk, t * 16, LDA, 0, lane);
      v16bf bk1 = load_frag_rowK(sk, t * 16, LDA, 32, lane);
      st[t] = WMMA_BF16(aq0, bk0, st[t]);
      st[t] = WMMA_BF16(aq1, bk1, st[t]);
    }

    // online softmax (row r = i + hi8; row spread over 16-lane half x 4 tiles)
    float alpha[8];
#pragma unroll
    for (int i = 0; i < 8; ++i) {
      float mx = -1e30f;
#pragma unroll
      for (int t = 0; t < 4; ++t) mx = fmaxf(mx, st[t][i] * scale);
#pragma unroll
      for (int m = 8; m >= 1; m >>= 1) mx = fmaxf(mx, __shfl_xor(mx, m, 32));
      float mnew = fmaxf(M[i], mx);
      alpha[i] = __expf(M[i] - mnew);
      float rs = 0.0f;
#pragma unroll
      for (int t = 0; t < 4; ++t) {
        float p = __expf(st[t][i] * scale - mnew);
        st[t][i] = p;
        rs += p;
      }
#pragma unroll
      for (int m = 8; m >= 1; m >>= 1) rs += __shfl_xor(rs, m, 32);
      L[i] = L[i] * alpha[i] + rs;
      M[i] = mnew;
    }

    // stage P (bf16) into per-wave LDS strip; rescale O
#pragma unroll
    for (int t = 0; t < 4; ++t) {
#pragma unroll
      for (int i = 0; i < 8; ++i) {
        sp[wv][(i + hi8) * LDA + t * 16 + ln] = f2bf(st[t][i]);
        o[t][i] *= alpha[i];
      }
    }
    asm volatile("s_wait_dscnt 0" ::: "memory");  // own strip: write->read ordering

    // O += P @ V
    v16bf pa0 = load_frag_rowK(sp[wv], 0, LDA, 0, lane);
    v16bf pa1 = load_frag_rowK(sp[wv], 0, LDA, 32, lane);
#pragma unroll
    for (int t = 0; t < 4; ++t) {
      v16bf bv0 = load_frag_rowK(svt, t * 16, LDA, 0, lane);
      v16bf bv1 = load_frag_rowK(svt, t * 16, LDA, 32, lane);
      o[t] = WMMA_BF16(pa0, bv0, o[t]);
      o[t] = WMMA_BF16(pa1, bv1, o[t]);
    }
  }

  const int b = bh >> 4, h = bh & 15;
#pragma unroll
  for (int t = 0; t < 4; ++t) {
#pragma unroll
    for (int i = 0; i < 8; ++i) {
      int srow = q0 + wv * 16 + i + hi8;
      int dcol = t * 16 + ln;
      out[((size_t)(b * S_ + srow)) * D_ + h * HD_ + dcol] = o[t][i] / L[i];
    }
  }
}

// ---------------- kernel 4: MoE (dense over experts, combine-weighted) ----------------
// grid (TOK/64, D/64), block 128 (4 waves)
__global__ void moe_gemm_kernel(const unsigned short* __restrict__ xb,
                                const unsigned short* __restrict__ webT,  // [E][N][K]
                                const float* __restrict__ bexp,
                                const float* __restrict__ comb,
                                float* __restrict__ out) {
  __shared__ __align__(16) unsigned short sx[64 * LDK];
  __shared__ __align__(16) unsigned short sw[64 * LDK];
  __shared__ float sg[64][E_];

  const int row0 = blockIdx.x * 64;
  const int col0 = blockIdx.y * 64;
  const int tid  = threadIdx.x;
  const int lane = tid & 31;
  const int wv   = tid >> 5;
  const int hi8  = (lane >> 4) * 8;
  const int ln   = lane & 15;

  for (int idx = tid; idx < 64 * E_; idx += 128) {
    int r = idx >> 3, e = idx & 7;
    sg[r][e] = comb[(size_t)(row0 + r) * E_ + e];
  }

  v8f tot[4];
#pragma unroll
  for (int t = 0; t < 4; ++t) tot[t] = vzero8();

  for (int e = 0; e < E_; ++e) {
    const unsigned short* we = webT + (size_t)e * D_ * D_;
    v8f pe[4];
#pragma unroll
    for (int t = 0; t < 4; ++t) pe[t] = vzero8();

    for (int k0 = 0; k0 < D_; k0 += 32) {
      __syncthreads();
      stage_tile<64, 4, LDK, 128>(sx, xb + (size_t)row0 * D_ + k0, D_, tid);
      stage_tile<64, 4, LDK, 128>(sw, we + (size_t)col0 * D_ + k0, D_, tid);
      __builtin_prefetch(we + (size_t)col0 * D_ + k0 + 32, 0, 1);
      async_wait_all();
      __syncthreads();

      v16bf a = load_frag_rowK(sx, wv * 16, LDK, 0, lane);
#pragma unroll
      for (int t = 0; t < 4; ++t) {
        v16bf b = load_frag_rowK(sw, t * 16, LDK, 0, lane);
        pe[t] = WMMA_BF16(a, b, pe[t]);
      }
    }

    // fold with per-row combine weight + expert bias
#pragma unroll
    for (int t = 0; t < 4; ++t) {
#pragma unroll
      for (int i = 0; i < 8; ++i) {
        int rl  = wv * 16 + i + hi8;
        int col = col0 + t * 16 + ln;
        float g = sg[rl][e];
        tot[t][i] += g * (pe[t][i] + bexp[e * D_ + col]);
      }
    }
  }

  // out += moe (attention already wrote out; stream-ordered)
#pragma unroll
  for (int t = 0; t < 4; ++t) {
#pragma unroll
    for (int i = 0; i < 8; ++i) {
      int rl  = wv * 16 + i + hi8;
      int col = col0 + t * 16 + ln;
      out[(size_t)(row0 + rl) * D_ + col] += tot[t][i];
    }
  }
}

// ---------------- launch ----------------
extern "C" void kernel_launch(void* const* d_in, const int* in_sizes, int n_in,
                              void* d_out, int out_size, void* d_ws, size_t ws_size,
                              hipStream_t stream) {
  const float* x    = (const float*)d_in[0];
  const float* Wqkv = (const float*)d_in[1];
  const float* bqkv = (const float*)d_in[2];
  const float* Wg   = (const float*)d_in[3];
  const float* bg   = (const float*)d_in[4];
  const float* Wex  = (const float*)d_in[5];
  const float* bex  = (const float*)d_in[6];
  float* out = (float*)d_out;

  char* ws = (char*)d_ws;
  size_t off = 0;
  unsigned short* xb     = (unsigned short*)(ws + off); off += (size_t)TOK_ * D_ * 2;
  unsigned short* wqkvT  = (unsigned short*)(ws + off); off += (size_t)D_ * N3D_ * 2;
  unsigned short* wexT   = (unsigned short*)(ws + off); off += (size_t)E_ * D_ * D_ * 2;
  unsigned short* qb     = (unsigned short*)(ws + off); off += (size_t)B_ * H_ * S_ * HD_ * 2;
  unsigned short* kb     = (unsigned short*)(ws + off); off += (size_t)B_ * H_ * S_ * HD_ * 2;
  unsigned short* vtb    = (unsigned short*)(ws + off); off += (size_t)B_ * H_ * S_ * HD_ * 2;
  float* comb            = (float*)(ws + off);          off += (size_t)TOK_ * E_ * 4;

  // 0) conversions: x -> bf16; weights -> bf16 TRANSPOSED ([n][k] row-major)
  cvt_bf16_kernel<<<1024, 256, 0, stream>>>(x, xb, (TOK_ * D_) / 4);
  cvt_bf16_transpose_kernel<<<dim3(D_ / 32, N3D_ / 32, 1), 256, 0, stream>>>(
      Wqkv, wqkvT, D_, N3D_);
  cvt_bf16_transpose_kernel<<<dim3(D_ / 32, D_ / 32, E_), 256, 0, stream>>>(
      Wex, wexT, D_, D_);

  // 1) QKV projection (q/k row-major heads, v transposed heads)
  qkv_gemm_kernel<<<dim3(TOK_ / 64, N3D_ / 64), 128, 0, stream>>>(
      xb, wqkvT, bqkv, qb, kb, vtb);

  // 2) gate + top-2 combine weights
  gate_topk_kernel<<<TOK_ / 8, 256, 0, stream>>>(x, Wg, bg, comb);

  // 3) flash attention -> writes d_out
  flash_attn_kernel<<<dim3(S_ / 64, B_ * H_), 128, 0, stream>>>(qb, kb, vtb, out);

  // 4) MoE -> accumulates into d_out
  moe_gemm_kernel<<<dim3(TOK_ / 64, D_ / 64), 128, 0, stream>>>(
      xb, wexT, bex, comb, out);
}
